// QuantizedLinear_55413668053588
// MI455X (gfx1250) — matrix operand order repaired and checked by reading, not compiled, not run on hardware
//
#include <hip/hip_runtime.h>

// CDNA5 (gfx1250) wave32 WMMA types
typedef __attribute__((ext_vector_type(16))) _Float16 v16h;
typedef __attribute__((ext_vector_type(8)))  _Float16 v8h;
typedef __attribute__((ext_vector_type(8)))  float    v8f;
typedef __attribute__((ext_vector_type(4)))  float    vf4;
typedef __attribute__((ext_vector_type(4)))  int      vi4;

#define K_DIM 4096
#define N_DIM 11008
#define M_DIM 64
#define KQ    1024   // K-range owned by each wave (K_DIM / 4 waves)

// ---------------------------------------------------------------------------
// Pre-pass: convert x (64x4096 f32) to f16 once. 1 MB read / 0.5 MB write.
// ---------------------------------------------------------------------------
__global__ __launch_bounds__(256)
void x_to_f16(const float* __restrict__ x, _Float16* __restrict__ xh)
{
    const int i = (blockIdx.x * 256 + threadIdx.x) * 8;
    vf4 a = *reinterpret_cast<const vf4*>(x + i);
    vf4 b = *reinterpret_cast<const vf4*>(x + i + 4);
    v8h h;
#pragma unroll
    for (int j = 0; j < 4; ++j) {
        h[j]     = (_Float16)a[j];
        h[4 + j] = (_Float16)b[j];
    }
    *reinterpret_cast<v8h*>(xh + i) = h;
}

// ---------------------------------------------------------------------------
// Main kernel. Block b owns output columns [16b, 16b+16). Wave w owns the
// K-range [1024w, 1024w+1024) and computes partial sums for ALL 64 output
// rows (4 accumulator tiles). Weights are therefore loaded from HBM exactly
// once and each int8->f16 B conversion feeds 4 WMMAs. A 16KB LDS reduction
// combines the 4 K-partials, then scale+bias are applied in f32.
// ---------------------------------------------------------------------------
template <bool PRECVT>
__global__ __launch_bounds__(128)
void QuantizedLinear_wmma_f16(const float*    __restrict__ xf,
                              const _Float16* __restrict__ xh,
                              const int*      __restrict__ wq,
                              const float*    __restrict__ scale_p,
                              const float*    __restrict__ bias,
                              float*          __restrict__ out)
{
    const int lane   = threadIdx.x & 31;
    const int wave   = threadIdx.x >> 5;      // 0..3 -> K quarter
    const int n_base = blockIdx.x << 4;       // 16 output columns per block
    const int g      = lane >> 4;             // half-wave group (0/1)
    const int ln     = lane & 15;             // 0..15
    const int kw     = wave * KQ;             // this wave's K offset

    // B-matrix (W^T) per-lane base: column n = n_base+ln; half g holds
    // K=k0+8g..k0+8g+7 and K=k0+16+8g..k0+16+8g+7 (matches A layout).
    const int* wB = wq + (size_t)(n_base + ln) * K_DIM + kw + (g << 3);

    v8f c0 = {}, c1 = {}, c2 = {}, c3 = {};

#pragma unroll 2
    for (int k0 = 0; k0 < KQ; k0 += 32) {
        // Prefetch the weight stream ~2KB ahead (global_prefetch_b8).
        __builtin_prefetch(wB + k0 + 512, 0, 0);

        // ---- load + convert B tile once (int32-stored int8 -> f16, exact) --
        const vi4* pb = reinterpret_cast<const vi4*>(wB + k0);
        vi4 b0 = pb[0], b1 = pb[1], b2 = pb[4], b3 = pb[5];
        v16h B;
#pragma unroll
        for (int i = 0; i < 4; ++i) {
            B[i]      = (_Float16)b0[i];
            B[4 + i]  = (_Float16)b1[i];
            B[8 + i]  = (_Float16)b2[i];
            B[12 + i] = (_Float16)b3[i];
        }

        // ---- load 4 A tiles (rows 16t..16t+15) and issue 4 WMMAs ----------
        v8f* acc[4] = { &c0, &c1, &c2, &c3 };
#pragma unroll
        for (int t = 0; t < 4; ++t) {
            v16h A;
            if (PRECVT) {
                // f16 x: lane needs K chunks {off..off+7} and {16+off..+7}
                const _Float16* pa =
                    xh + (size_t)((t << 4) + ln) * K_DIM + kw + k0 + (g << 3);
                v8h lo = *reinterpret_cast<const v8h*>(pa);
                v8h hi = *reinterpret_cast<const v8h*>(pa + 16);
                A = __builtin_shufflevector(lo, hi, 0, 1, 2, 3, 4, 5, 6, 7,
                                            8, 9, 10, 11, 12, 13, 14, 15);
            } else {
                const float* pa =
                    xf + (size_t)((t << 4) + ln) * K_DIM + kw + k0 + (g << 3);
                vf4 a0 = *reinterpret_cast<const vf4*>(pa);
                vf4 a1 = *reinterpret_cast<const vf4*>(pa + 4);
                vf4 a2 = *reinterpret_cast<const vf4*>(pa + 16);
                vf4 a3 = *reinterpret_cast<const vf4*>(pa + 20);
#pragma unroll
                for (int i = 0; i < 4; ++i) {
                    A[i]      = (_Float16)a0[i];
                    A[4 + i]  = (_Float16)a1[i];
                    A[8 + i]  = (_Float16)a2[i];
                    A[12 + i] = (_Float16)a3[i];
                }
            }
            *acc[t] = __builtin_amdgcn_wmma_f32_16x16x32_f16(
                          false, A, false, B, (short)0, *acc[t], false, false);
        }
    }

    // ---- LDS reduction of the 4 K-partials --------------------------------
    __shared__ v8f red[4][4][32];      // [producer wave][tile][lane] = 16 KB
    red[wave][0][lane] = c0;
    red[wave][1][lane] = c1;
    red[wave][2][lane] = c2;
    red[wave][3][lane] = c3;
    __syncthreads();

    // Wave w reduces tile t = w (rows 16w..16w+15).
    v8f acc = red[0][wave][lane];
    acc += red[1][wave][lane];
    acc += red[2][wave][lane];
    acc += red[3][wave][lane];

    const float s  = scale_p[0];
    const float bv = bias[n_base + ln];
#pragma unroll
    for (int r = 0; r < 8; ++r) {
        const int m = (wave << 4) + r + (g << 3);   // VGPR r, half-wave g
        out[(size_t)m * N_DIM + n_base + ln] = acc[r] * s + bv;
    }
}

extern "C" void kernel_launch(void* const* d_in, const int* in_sizes, int n_in,
                              void* d_out, int out_size, void* d_ws, size_t ws_size,
                              hipStream_t stream) {
    const float* x     = (const float*)d_in[0];
    const int*   wq    = (const int*)  d_in[1];
    const float* scale = (const float*)d_in[2];
    const float* bias  = (const float*)d_in[3];
    float*       out   = (float*)d_out;

    const dim3 grid(N_DIM / 16, 1, 1);   // 688 blocks, one 16-col strip each
    const dim3 block(128, 1, 1);         // 4 waves: K quarters 0..3

    const size_t need = (size_t)M_DIM * K_DIM * sizeof(_Float16);  // 512 KB
    if (ws_size >= need) {
        _Float16* xh = (_Float16*)d_ws;
        // 64*4096 / (256 threads * 8 elem) = 128 blocks
        x_to_f16<<<dim3(128), dim3(256), 0, stream>>>(x, xh);
        QuantizedLinear_wmma_f16<true>
            <<<grid, block, 0, stream>>>(x, xh, wq, scale, bias, out);
    } else {
        QuantizedLinear_wmma_f16<false>
            <<<grid, block, 0, stream>>>(x, nullptr, wq, scale, bias, out);
    }
}
